// MoELinear_29042568855724
// MI455X (gfx1250) — compile-verified, matrix operand order checked by reading, not run on hardware
//
#include <hip/hip_runtime.h>
#include <hip/hip_bf16.h>
#include <math.h>

// Problem constants (from reference)
#define T_TOK 8192
#define DIN   1024
#define DOUT  1024
#define DPROJ 2048
#define NE    8

typedef __bf16 bf16;
typedef __attribute__((ext_vector_type(16))) __bf16 v16bf;
typedef __attribute__((ext_vector_type(8)))  __bf16 v8bf;
typedef __attribute__((ext_vector_type(8)))  float  v8f;

union V16 { v16bf v; v8bf h[2]; };

struct __align__(8) B4 { bf16 a, b, c, d; };

// GEMM tiling
#define TM 128
#define TN 128
#define TK 64    // staged K per sync stage (2 x K=32 WMMA sub-steps)
#define LDT 72   // padded LDS row length (elements); 144B rows: 16B-aligned, conflict-free

// Workspace layout (bytes)
#define OFF_XB    ((size_t)0)                     // bf16 x            : 16 MB
#define OFF_W1T   ((size_t)16777216)              // bf16 W1^T [E][P][I]: 32 MB
#define OFF_W2T   ((size_t)50331648)              // bf16 W2^T [E][O][P]: 32 MB
#define OFF_H     ((size_t)83886080)              // bf16 H [2T][PROJ]  : 64 MB
#define OFF_IDX   ((size_t)150994944)             // int  idxList [E][T]
#define OFF_GATE  ((size_t)151257088)             // f32  gateList[E][T]
#define OFF_EID   ((size_t)151519232)             // int  eid [T][2]
#define OFF_GW    ((size_t)151584768)             // f32  gw  [T][2]
#define OFF_CNT   ((size_t)151650304)             // int  counts[E]
#define OFF_HOFF  ((size_t)151650336)             // int  hoff[E]

__device__ __forceinline__ float gelu_tanh(float x) {
  float x3 = x * x * x;
  return 0.5f * x * (1.0f + tanhf(0.7978845608028654f * (x + 0.044715f * x3)));
}

// Issue one 16-byte async global->LDS copy (CDNA5, tracked by ASYNCcnt).
__device__ __forceinline__ void async_copy_b128(const bf16* gsrc, unsigned lds_off) {
  unsigned long long ga = (unsigned long long)(size_t)gsrc;
  asm volatile("global_load_async_to_lds_b128 %0, %1, off"
               :: "v"(lds_off), "v"(ga) : "memory");
}

__device__ __forceinline__ void wait_async0() {
  asm volatile("s_wait_asynccnt 0x0" ::: "memory");
}

// ---------------- f32 -> bf16 elementwise ----------------
__global__ __launch_bounds__(256) void convert_x_kernel(const float* __restrict__ x,
                                                        bf16* __restrict__ xb) {
  int i = (blockIdx.x * 256 + threadIdx.x) * 4;
  float4 v = *(const float4*)(x + i);
  B4 o; o.a = (bf16)v.x; o.b = (bf16)v.y; o.c = (bf16)v.z; o.d = (bf16)v.w;
  *(B4*)(xb + i) = o;
}

// ------------- per-expert transpose + convert ------------
// src: [E][R][C] f32, dst: [E][C][R] bf16; block (32,8), grid (C/32, R/32, E)
__global__ __launch_bounds__(256) void transpose_convert_kernel(const float* __restrict__ src,
                                                                bf16* __restrict__ dst,
                                                                int R, int C) {
  __shared__ float tile[32][33];
  int e = blockIdx.z;
  const float* s = src + (size_t)e * R * C;
  bf16* d = dst + (size_t)e * R * C;
  int c0 = blockIdx.x * 32, r0 = blockIdx.y * 32;
  for (int j = 0; j < 4; ++j) {
    int r = threadIdx.y + j * 8;
    tile[r][threadIdx.x] = s[(size_t)(r0 + r) * C + c0 + threadIdx.x];
  }
  __syncthreads();
  for (int j = 0; j < 4; ++j) {
    int c = threadIdx.y + j * 8;
    d[(size_t)(c0 + c) * R + r0 + threadIdx.x] = (bf16)tile[threadIdx.x][c];
  }
}

// ---------------- gating: one wave32 per token ----------------
__global__ __launch_bounds__(256) void gate_kernel(const float* __restrict__ x,
                                                   const float* __restrict__ Wg,
                                                   const float* __restrict__ bg,
                                                   int* __restrict__ eid,
                                                   float* __restrict__ gw) {
  int wave = threadIdx.x >> 5, lane = threadIdx.x & 31;
  int t = blockIdx.x * 8 + wave;
  float acc[NE];
  for (int e = 0; e < NE; ++e) acc[e] = 0.0f;
  const float* xr = x + (size_t)t * DIN;
  for (int i = lane; i < DIN; i += 32) {
    float xv = xr[i];
    const float4* w = (const float4*)(Wg + i * NE);
    float4 w0 = w[0], w1 = w[1];
    acc[0] += xv * w0.x; acc[1] += xv * w0.y; acc[2] += xv * w0.z; acc[3] += xv * w0.w;
    acc[4] += xv * w1.x; acc[5] += xv * w1.y; acc[6] += xv * w1.z; acc[7] += xv * w1.w;
  }
  for (int e = 0; e < NE; ++e)
    for (int off = 16; off > 0; off >>= 1)
      acc[e] += __shfl_xor(acc[e], off, 32);
  if (lane == 0) {
    float l[NE];
    for (int e = 0; e < NE; ++e) l[e] = (acc[e] + bg[e]) * (1.0f / 32.0f);  // /sqrt(1024)
    int i1 = 0; float v1 = l[0];
    for (int e = 1; e < NE; ++e) if (l[e] > v1) { v1 = l[e]; i1 = e; }
    int i2 = (i1 == 0) ? 1 : 0; float v2 = l[i2];
    for (int e = 0; e < NE; ++e) if (e != i1 && l[e] > v2) { v2 = l[e]; i2 = e; }
    float denom = 0.0f;
    for (int e = 0; e < NE; ++e) if (l[e] >= v2) denom += __expf(l[e] - v1);
    eid[t * 2 + 0] = i1; eid[t * 2 + 1] = i2;
    gw[t * 2 + 0] = 1.0f / denom;
    gw[t * 2 + 1] = __expf(v2 - v1) / denom;
  }
}

// --------- deterministic per-expert compaction (one wave per expert) ---------
__global__ void compact_kernel(const int* __restrict__ eid, const float* __restrict__ gw,
                               int* __restrict__ idxList, float* __restrict__ gateList,
                               int* __restrict__ counts) {
  int e = blockIdx.x, lane = threadIdx.x;
  int base = 0;
  for (int start = 0; start < T_TOK; start += 32) {
    int t = start + lane;
    int e0 = eid[t * 2], e1 = eid[t * 2 + 1];
    bool pred = (e0 == e) || (e1 == e);
    float g = (e0 == e) ? gw[t * 2] : gw[t * 2 + 1];
    unsigned mlo = (unsigned)__ballot(pred);
    int pos = base + __popc(mlo & ((1u << lane) - 1u));
    if (pred) { idxList[e * T_TOK + pos] = t; gateList[e * T_TOK + pos] = g; }
    base += __popc(mlo);
  }
  if (lane == 0) counts[e] = base;
}

__global__ void offsets_kernel(const int* __restrict__ counts, int* __restrict__ hoff) {
  int run = 0;
  for (int e = 0; e < NE; ++e) { hoff[e] = run; run += counts[e]; }
}

__global__ __launch_bounds__(256) void zero_out_kernel(float* __restrict__ out) {
  int i = (blockIdx.x * 256 + threadIdx.x) * 4;
  *(float4*)(out + i) = make_float4(0.f, 0.f, 0.f, 0.f);
}

// ------------- GEMM1: H = gelu(Xg @ W1^T_e + b1_e), gathered rows -------------
__global__ __launch_bounds__(256, 1) void gemm1_kernel(const bf16* __restrict__ xb,
                                                       const bf16* __restrict__ w1t,
                                                       const float* __restrict__ b1,
                                                       const int* __restrict__ idxList,
                                                       const int* __restrict__ counts,
                                                       const int* __restrict__ hoff,
                                                       bf16* __restrict__ H) {
  int e = blockIdx.z;
  int cnt = counts[e];
  int mtile = blockIdx.y;
  if (mtile * TM >= cnt) return;
  int ntile = blockIdx.x;

  __shared__ __align__(16) bf16 As[2][TM * LDT];
  __shared__ __align__(16) bf16 Bs[2][TN * LDT];
  __shared__ int tok[TM];

  int tid = threadIdx.x;
  if (tid < TM) {
    int r = mtile * TM + tid;
    tok[tid] = idxList[e * T_TOK + (r < cnt ? r : 0)];
  }
  __syncthreads();

  const bf16* Bbase = w1t + ((size_t)e * DPROJ + (size_t)ntile * TN) * DIN;

  // Loop-invariant async-copy addressing: 4 x 16B segments per operand per thread.
  const bf16* aG[4]; const bf16* bG[4]; unsigned aL[4], bL[4];
#pragma unroll
  for (int s = 0; s < 4; ++s) {
    int seg = tid + s * 256;                 // 1024 segments: 128 rows x 8 cols of 16B
    int row = seg >> 3, sc = (seg & 7) * 8;
    aG[s] = xb + (size_t)tok[row] * DIN + sc;
    bG[s] = Bbase + (size_t)row * DIN + sc;
    aL[s] = (unsigned)(size_t)&As[0][row * LDT + sc];
    bL[s] = (unsigned)(size_t)&Bs[0][row * LDT + sc];
  }
  const unsigned bufStride = (unsigned)(TM * LDT * sizeof(bf16));

  int wave = tid >> 5, lane = tid & 31;
  int waveM = wave & 3, waveN = wave >> 2;   // 4 x 2 wave grid; wave tile 32(M) x 64(N)
  int m16 = lane & 15, half = lane >> 4;

  v8f acc[2][4] = {};

  auto issue_tile = [&](int buf, int k0) {
    unsigned d = buf ? bufStride : 0u;
#pragma unroll
    for (int s = 0; s < 4; ++s) {
      async_copy_b128(aG[s] + k0, aL[s] + d);
      async_copy_b128(bG[s] + k0, bL[s] + d);
    }
  };

  const int nk = DIN / TK;
  issue_tile(0, 0);
  for (int ki = 0; ki < nk; ++ki) {
    wait_async0();          // my async copies for buffer ki&1 have landed
    __syncthreads();        // everyone's landed; everyone's reads of other buffer retired
    if (ki + 1 < nk) issue_tile((ki + 1) & 1, (ki + 1) * TK);
    int buf = ki & 1;
#pragma unroll
    for (int h = 0; h < 2; ++h) {            // two K=32 sub-steps per staged chunk
      V16 a[2], b[4];
#pragma unroll
      for (int mi = 0; mi < 2; ++mi) {
        const bf16* p = &As[buf][(waveM * 32 + mi * 16 + m16) * LDT + h * 32 + 8 * half];
        a[mi].h[0] = *(const v8bf*)p;         // K = 8*half .. +7
        a[mi].h[1] = *(const v8bf*)(p + 16);  // K = 16+8*half .. +7
      }
#pragma unroll
      for (int ni = 0; ni < 4; ++ni) {
        const bf16* p = &Bs[buf][(waveN * 64 + ni * 16 + m16) * LDT + h * 32 + 8 * half];
        b[ni].h[0] = *(const v8bf*)p;
        b[ni].h[1] = *(const v8bf*)(p + 16);
      }
#pragma unroll
      for (int mi = 0; mi < 2; ++mi)
#pragma unroll
        for (int ni = 0; ni < 4; ++ni)
          acc[mi][ni] = __builtin_amdgcn_wmma_f32_16x16x32_bf16(
              false, a[mi].v, false, b[ni].v, (short)0, acc[mi][ni], false, false);
    }
  }

  int hbase = hoff[e] + mtile * TM;
#pragma unroll
  for (int mi = 0; mi < 2; ++mi)
#pragma unroll
    for (int ni = 0; ni < 4; ++ni) {
      int col = ntile * TN + waveN * 64 + ni * 16 + m16;
      float bias = b1[e * DPROJ + col];
#pragma unroll
      for (int r = 0; r < 8; ++r) {
        int mrow = waveM * 32 + mi * 16 + r + 8 * half;
        if (mtile * TM + mrow < cnt) {
          float v = gelu_tanh(acc[mi][ni][r] + bias);
          H[(size_t)(hbase + mrow) * DPROJ + col] = (bf16)v;
        }
      }
    }
}

// -------- GEMM2: out[tok] += gate * (H @ W2^T_e + b2_e)  (atomic scatter) --------
__global__ __launch_bounds__(256, 1) void gemm2_kernel(const bf16* __restrict__ H,
                                                       const bf16* __restrict__ w2t,
                                                       const float* __restrict__ b2,
                                                       const int* __restrict__ idxList,
                                                       const float* __restrict__ gateList,
                                                       const int* __restrict__ counts,
                                                       const int* __restrict__ hoff,
                                                       float* __restrict__ out) {
  int e = blockIdx.z;
  int cnt = counts[e];
  int mtile = blockIdx.y;
  if (mtile * TM >= cnt) return;
  int ntile = blockIdx.x;

  __shared__ __align__(16) bf16 As[2][TM * LDT];
  __shared__ __align__(16) bf16 Bs[2][TN * LDT];
  __shared__ int tok[TM];
  __shared__ float gts[TM];

  int tid = threadIdx.x;
  int hbase = hoff[e];
  if (tid < TM) {
    int r = mtile * TM + tid;
    int rc = r < cnt ? r : 0;
    tok[tid] = idxList[e * T_TOK + rc];
    gts[tid] = gateList[e * T_TOK + rc];
  }
  __syncthreads();

  const bf16* Bbase = w2t + ((size_t)e * DOUT + (size_t)ntile * TN) * DPROJ;

  const bf16* aG[4]; const bf16* bG[4]; unsigned aL[4], bL[4];
#pragma unroll
  for (int s = 0; s < 4; ++s) {
    int seg = tid + s * 256;
    int row = seg >> 3, sc = (seg & 7) * 8;
    int rr = mtile * TM + row;
    if (rr >= cnt) rr = cnt - 1;
    aG[s] = H + (size_t)(hbase + rr) * DPROJ + sc;
    bG[s] = Bbase + (size_t)row * DPROJ + sc;
    aL[s] = (unsigned)(size_t)&As[0][row * LDT + sc];
    bL[s] = (unsigned)(size_t)&Bs[0][row * LDT + sc];
  }
  const unsigned bufStride = (unsigned)(TM * LDT * sizeof(bf16));

  int wave = tid >> 5, lane = tid & 31;
  int waveM = wave & 3, waveN = wave >> 2;
  int m16 = lane & 15, half = lane >> 4;

  v8f acc[2][4] = {};

  auto issue_tile = [&](int buf, int k0) {
    unsigned d = buf ? bufStride : 0u;
#pragma unroll
    for (int s = 0; s < 4; ++s) {
      async_copy_b128(aG[s] + k0, aL[s] + d);
      async_copy_b128(bG[s] + k0, bL[s] + d);
    }
  };

  const int nk = DPROJ / TK;
  issue_tile(0, 0);
  for (int ki = 0; ki < nk; ++ki) {
    wait_async0();
    __syncthreads();
    if (ki + 1 < nk) issue_tile((ki + 1) & 1, (ki + 1) * TK);
    int buf = ki & 1;
#pragma unroll
    for (int h = 0; h < 2; ++h) {
      V16 a[2], b[4];
#pragma unroll
      for (int mi = 0; mi < 2; ++mi) {
        const bf16* p = &As[buf][(waveM * 32 + mi * 16 + m16) * LDT + h * 32 + 8 * half];
        a[mi].h[0] = *(const v8bf*)p;
        a[mi].h[1] = *(const v8bf*)(p + 16);
      }
#pragma unroll
      for (int ni = 0; ni < 4; ++ni) {
        const bf16* p = &Bs[buf][(waveN * 64 + ni * 16 + m16) * LDT + h * 32 + 8 * half];
        b[ni].h[0] = *(const v8bf*)p;
        b[ni].h[1] = *(const v8bf*)(p + 16);
      }
#pragma unroll
      for (int mi = 0; mi < 2; ++mi)
#pragma unroll
        for (int ni = 0; ni < 4; ++ni)
          acc[mi][ni] = __builtin_amdgcn_wmma_f32_16x16x32_bf16(
              false, a[mi].v, false, b[ni].v, (short)0, acc[mi][ni], false, false);
    }
  }

#pragma unroll
  for (int mi = 0; mi < 2; ++mi)
#pragma unroll
    for (int ni = 0; ni < 4; ++ni) {
      int col = ntile * TN + waveN * 64 + ni * 16 + m16;
      float bias = b2[e * DOUT + col];
#pragma unroll
      for (int r = 0; r < 8; ++r) {
        int mrow = waveM * 32 + mi * 16 + r + 8 * half;
        if (mtile * TM + mrow < cnt) {
          float v = (acc[mi][ni][r] + bias) * gts[mrow];
          atomicAdd(out + (size_t)tok[mrow] * DOUT + col, v);
        }
      }
    }
}

extern "C" void kernel_launch(void* const* d_in, const int* in_sizes, int n_in,
                              void* d_out, int out_size, void* d_ws, size_t ws_size,
                              hipStream_t stream) {
  const float* x  = (const float*)d_in[0];
  const float* Wg = (const float*)d_in[1];
  const float* bg = (const float*)d_in[2];
  const float* W1 = (const float*)d_in[3];
  const float* b1 = (const float*)d_in[4];
  const float* W2 = (const float*)d_in[5];
  const float* b2 = (const float*)d_in[6];
  float* out = (float*)d_out;

  char* ws = (char*)d_ws;
  bf16*  xb       = (bf16*)(ws + OFF_XB);
  bf16*  w1t      = (bf16*)(ws + OFF_W1T);
  bf16*  w2t      = (bf16*)(ws + OFF_W2T);
  bf16*  Hbuf     = (bf16*)(ws + OFF_H);
  int*   idxList  = (int*)(ws + OFF_IDX);
  float* gateList = (float*)(ws + OFF_GATE);
  int*   eid      = (int*)(ws + OFF_EID);
  float* gwbuf    = (float*)(ws + OFF_GW);
  int*   counts   = (int*)(ws + OFF_CNT);
  int*   hoff     = (int*)(ws + OFF_HOFF);

  // 1) precision/layout conversion
  convert_x_kernel<<<(T_TOK * DIN) / 1024, 256, 0, stream>>>(x, xb);
  transpose_convert_kernel<<<dim3(DPROJ / 32, DIN / 32, NE), dim3(32, 8), 0, stream>>>(
      W1, w1t, DIN, DPROJ);
  transpose_convert_kernel<<<dim3(DOUT / 32, DPROJ / 32, NE), dim3(32, 8), 0, stream>>>(
      W2, w2t, DPROJ, DOUT);

  // 2) routing
  gate_kernel<<<T_TOK / 8, 256, 0, stream>>>(x, Wg, bg, eid, gwbuf);
  compact_kernel<<<NE, 32, 0, stream>>>(eid, gwbuf, idxList, gateList, counts);
  offsets_kernel<<<1, 1, 0, stream>>>(counts, hoff);

  // 3) expert compute
  zero_out_kernel<<<(T_TOK * DOUT) / 1024, 256, 0, stream>>>(out);
  gemm1_kernel<<<dim3(DPROJ / TN, T_TOK / TM, NE), 256, 0, stream>>>(
      xb, w1t, b1, idxList, counts, hoff, Hbuf);
  gemm2_kernel<<<dim3(DOUT / TN, T_TOK / TM, NE), 256, 0, stream>>>(
      Hbuf, w2t, b2, idxList, gateList, counts, hoff, out);
}
